// PointMisFeat_88527865905289
// MI455X (gfx1250) — matrix operand order repaired and checked by reading, not compiled, not run on hardware
//
#include <hip/hip_runtime.h>
#include <hip/hip_bf16.h>
#include <math.h>

// Problem constants (from reference setup_inputs)
#define B_   4
#define N_   8192
#define M_   2048
#define KNN_ 16
#define P_   (KNN_ * M_)       // 32768 pixels per batch (p = k*M + m)
#define TPB_ (P_ / 16)         // 2048 16-pixel tiles per batch

typedef __attribute__((ext_vector_type(16))) __bf16 v16bf;
typedef __attribute__((ext_vector_type(8)))  __bf16 v8bf;
typedef __attribute__((ext_vector_type(8)))  float  v8f;

// ---------------------------------------------------------------------------
// Packed B-fragment activation index:  [b][tile][kt][lane][e]
//   k-tile kt = c>>5, lane = 16*((c>>4)&1) + (p&15), e = c&15
// Fragments are the per-lane v16bf fed to WMMA B; k = 32*kt + 16*half + e, same order as A's packing.
// ---------------------------------------------------------------------------
__device__ __forceinline__ size_t pidx(int b, int c, int p, int nkt) {
  int tile = p >> 4, col = p & 15;
  int kt   = c >> 5;
  int lane = (((c >> 4) & 1) << 4) + col;
  int e    = c & 15;
  return ((((size_t)b * TPB_ + tile) * nkt + kt) * 32 + lane) * 16 + e;
}

// ---------------------------------------------------------------------------
// Float atomic max (lowers to global_atomic_max_num_f32 on gfx1250)
// ---------------------------------------------------------------------------
__device__ __forceinline__ void atomic_max_f32(float* addr, float val) {
#if defined(__has_builtin) && __has_builtin(__hip_atomic_fetch_max)
  __hip_atomic_fetch_max(addr, val, __ATOMIC_RELAXED, __HIP_MEMORY_SCOPE_AGENT);
#else
  int old = __float_as_int(*addr);
  while (val > __int_as_float(old)) {
    int assumed = old;
    old = atomicCAS((int*)addr, assumed, __float_as_int(val));
    if (old == assumed) break;
  }
#endif
}

// ---------------------------------------------------------------------------
// Init output to -inf (final reduction is an atomic max)
// ---------------------------------------------------------------------------
__global__ __launch_bounds__(256) void init_out_kernel(float* out, int n) {
  int i = blockIdx.x * 256 + threadIdx.x;
  if (i < n) out[i] = -INFINITY;
}

// ---------------------------------------------------------------------------
// Pack one f32 weight matrix (O x CR, row-major) into bf16 A-fragment order:
//   [ot][kt][lane][e];  row = lane&15, half = lane>>4,
//   K = kt*32 + half*16 + e;  zero-padded for K >= CR.
// ---------------------------------------------------------------------------
__global__ __launch_bounds__(256) void pack_weights_kernel(const float* __restrict__ W,
                                                           __bf16* __restrict__ Wp,
                                                           int O, int CR, int NKT) {
  int i = blockIdx.x * 256 + threadIdx.x;
  int total = (O >> 4) * NKT * 512;
  if (i >= total) return;
  int e    = i & 15;
  int lane = (i >> 4) & 31;
  int kt   = (i >> 9) % NKT;
  int ot   = (i >> 9) / NKT;
  int row  = lane & 15;
  int half = lane >> 4;
  int K    = kt * 32 + half * 16 + e;
  float v  = (K < CR) ? W[(size_t)(ot * 16 + row) * CR + K] : 0.0f;
  Wp[i] = (__bf16)v;
}

// ---------------------------------------------------------------------------
// KNN + feature build.  One 256-thread block per (b, m) query.
// Distances for all N points staged in LDS (32KB), then 16 argmin rounds.
// Writes bf16 features packed B-fragment layout, channels 10..31 zeroed.
// ---------------------------------------------------------------------------
__global__ __launch_bounds__(256) void knn_feat_kernel(const float* __restrict__ xyz,
                                                       const float* __restrict__ mis,
                                                       __bf16* __restrict__ feat) {
  __shared__ float dist[N_];
  __shared__ float redv[256];
  __shared__ int   redi[256];
  __shared__ int   selIdx[KNN_];

  const int tid = threadIdx.x;
  const int b = blockIdx.x / M_;
  const int m = blockIdx.x % M_;

  const float qx = mis[((size_t)b * 3 + 0) * M_ + m];
  const float qy = mis[((size_t)b * 3 + 1) * M_ + m];
  const float qz = mis[((size_t)b * 3 + 2) * M_ + m];
  const float* xb = xyz + (size_t)b * 3 * N_;

  for (int j = tid; j < N_; j += 256) {
    float dx = xb[j]        - qx;
    float dy = xb[N_ + j]   - qy;
    float dz = xb[2*N_ + j] - qz;
    dist[j] = dx * dx + dy * dy + dz * dz;
  }
  __syncthreads();

  for (int it = 0; it < KNN_; ++it) {
    float best = INFINITY; int bi = N_;
    for (int j = tid; j < N_; j += 256) {
      float v = dist[j];
      if (v < best) { best = v; bi = j; }
    }
    redv[tid] = best; redi[tid] = bi;
    __syncthreads();
    for (int s = 128; s > 0; s >>= 1) {
      if (tid < s) {
        float v2 = redv[tid + s]; int i2 = redi[tid + s];
        if (v2 < redv[tid] || (v2 == redv[tid] && i2 < redi[tid])) {
          redv[tid] = v2; redi[tid] = i2;
        }
      }
      __syncthreads();
    }
    if (tid == 0) { selIdx[it] = redi[0]; dist[redi[0]] = INFINITY; }
    __syncthreads();
  }

  // 10 real channels (rel, grouped, center, dist), padded to 32 with zeros
  for (int idx = tid; idx < 32 * KNN_; idx += 256) {
    int c = idx >> 4;
    int k = idx & 15;
    float val = 0.f;
    if (c < 10) {
      int j = selIdx[k];
      float gx = xb[j], gy = xb[N_ + j], gz = xb[2*N_ + j];
      float rx = gx - qx, ry = gy - qy, rz = gz - qz;
      if      (c == 0) val = rx;
      else if (c == 1) val = ry;
      else if (c == 2) val = rz;
      else if (c == 3) val = gx;
      else if (c == 4) val = gy;
      else if (c == 5) val = gz;
      else if (c == 6) val = qx;
      else if (c == 7) val = qy;
      else if (c == 8) val = qz;
      else             val = sqrtf(rx*rx + ry*ry + rz*rz + 1e-12f);
    }
    feat[pidx(b, c, k * M_ + m, 1)] = (__bf16)val;
  }
}

// ---------------------------------------------------------------------------
// GEMM over register-resident X fragments (A from packed global weights)
// ---------------------------------------------------------------------------
template<int NKT>
__device__ __forceinline__ v8f gemm_reg(const v16bf* xf, const __bf16* __restrict__ Wp,
                                        int ot, v8f acc) {
  const int lane = threadIdx.x & 31;
#pragma unroll
  for (int kt = 0; kt < NKT; ++kt) {
    v16bf a = *(const v16bf*)(Wp + ((size_t)(ot * NKT + kt) * 32 + lane) * 16);
    acc = __builtin_amdgcn_wmma_f32_16x16x32_bf16(false, a, false, xf[kt],
                                                  (short)0, acc, false, false);
  }
  return acc;
}

// GEMM over LDS-staged (packed) activations
template<int NKT>
__device__ __forceinline__ v8f gemm_lds(const __bf16* S, const __bf16* __restrict__ Wp,
                                        int ot, v8f acc) {
  const int lane = threadIdx.x & 31;
#pragma unroll
  for (int kt = 0; kt < NKT; ++kt) {
    v16bf a = *(const v16bf*)(Wp + ((size_t)(ot * NKT + kt) * 32 + lane) * 16);
    v16bf bf = *(const v16bf*)(S + (size_t)kt * 512 + lane * 16);
    acc = __builtin_amdgcn_wmma_f32_16x16x32_bf16(false, a, false, bf,
                                                  (short)0, acc, false, false);
  }
  return acc;
}

__device__ __forceinline__ v8f bias_tile(const float* __restrict__ b, int o0) {
  const int lane  = threadIdx.x & 31;
  const int rbase = (lane >> 4) * 8;
  v8f acc;
#pragma unroll
  for (int r = 0; r < 8; ++r) acc[r] = b[o0 + rbase + r];
  return acc;
}

// Convert D tile (v8f) to 8 packed bf16 (one contiguous fragment slice)
template<bool LEAKY>
__device__ __forceinline__ v8bf pack_tile(v8f acc) {
  v8bf pk;
#pragma unroll
  for (int r = 0; r < 8; ++r) {
    float v = acc[r];
    if (LEAKY) v = (v >= 0.f) ? v : 0.2f * v;
    pk[r] = (__bf16)v;
  }
  return pk;
}

// D-tile scatter into packed B-fragment LDS: one ds_store_b128 per lane.
// Channel k = o0 + 8*half + r  ->  e = 8*half + r (8 consecutive slots).
template<bool LEAKY>
__device__ __forceinline__ void store_tile_lds(__bf16* Hs, int o0, v8f acc) {
  const int lane  = threadIdx.x & 31;
  const int col   = lane & 15;
  const int rbase = (lane >> 4) * 8;
  v8bf pk = pack_tile<LEAKY>(acc);
  const int lane2 = (((o0 >> 4) & 1) << 4) + col;
  *(v8bf*)(Hs + ((size_t)(o0 >> 5) * 32 + lane2) * 16 + rbase) = pk;
}

// ---------------------------------------------------------------------------
// Fused residual block: one wave32 per 16-pixel tile.
// out = W3*leaky(W2*leaky(W1*x+b1)+b2)+b3 + (Wsc*x+bsc)
// X fragments live in registers; h1/h2 staged in LDS in fragment order
// (same-wave DS ops are in-order, no barriers).  FINAL=true fuses the
// max-over-K reduction via global float atomic max.
// ---------------------------------------------------------------------------
template<int NKT_X, int H1, int H2, int OUTC, bool FINAL>
__global__ __launch_bounds__(32) void res_block_kernel(
    const __bf16* __restrict__ Xp,
    const __bf16* __restrict__ Wsc_p, const float* __restrict__ bsc,
    const __bf16* __restrict__ W1_p,  const float* __restrict__ b1,
    const __bf16* __restrict__ W2_p,  const float* __restrict__ b2,
    const __bf16* __restrict__ W3_p,  const float* __restrict__ b3,
    __bf16* __restrict__ Yp, float* __restrict__ outF) {
  constexpr int NKT_H1 = H1 / 32;
  constexpr int NKT_H2 = H2 / 32;
  __shared__ __bf16 H1s[H1 * 16];
  __shared__ __bf16 H2s[H2 * 16];

  const int lane = threadIdx.x;
  const int tile = blockIdx.x;                 // b * TPB_ + tileInBatch
  const int b    = tile / TPB_;
  const int p0   = (tile % TPB_) * 16;

  // X fragments -> registers (coalesced 32B per lane per k-tile)
  v16bf xf[NKT_X];
#pragma unroll
  for (int kt = 0; kt < NKT_X; ++kt)
    xf[kt] = *(const v16bf*)(Xp + ((size_t)tile * NKT_X + kt) * 512 + lane * 16);

#pragma unroll
  for (int o0 = 0; o0 < H1; o0 += 16) {
    v8f acc = bias_tile(b1, o0);
    acc = gemm_reg<NKT_X>(xf, W1_p, o0 >> 4, acc);
    store_tile_lds<true>(H1s, o0, acc);
  }
#pragma unroll
  for (int o0 = 0; o0 < H2; o0 += 16) {
    v8f acc = bias_tile(b2, o0);
    acc = gemm_lds<NKT_H1>(H1s, W2_p, o0 >> 4, acc);
    store_tile_lds<true>(H2s, o0, acc);
  }

  const int col   = lane & 15;
  const int rbase = (lane >> 4) * 8;
#pragma unroll
  for (int o0 = 0; o0 < OUTC; o0 += 16) {
    v8f acc   = bias_tile(b3,  o0);
    v8f accsc = bias_tile(bsc, o0);
    acc   = gemm_lds<NKT_H2>(H2s, W3_p,  o0 >> 4, acc);
    accsc = gemm_reg<NKT_X> (xf,  Wsc_p, o0 >> 4, accsc);
#pragma unroll
    for (int r = 0; r < 8; ++r) acc[r] += accsc[r];
    if (FINAL) {
      const int p = p0 + col;
      const int m = p & (M_ - 1);
#pragma unroll
      for (int r = 0; r < 8; ++r)
        atomic_max_f32(&outF[((size_t)b * OUTC + o0 + rbase + r) * M_ + m], acc[r]);
    } else {
      // packed global store: one global_store_b128 per lane per o-tile
      v8bf pk = pack_tile<false>(acc);
      const int lane2 = (((o0 >> 4) & 1) << 4) + col;
      *(v8bf*)(Yp + (((size_t)tile * (OUTC / 32) + (o0 >> 5)) * 32 + lane2) * 16 + rbase) = pk;
    }
  }
}

// ---------------------------------------------------------------------------
// x1 = max over K, concat [x, broadcast(x1)] -> 128 channels (packed bf16)
// ---------------------------------------------------------------------------
__global__ __launch_bounds__(256) void concat_max_kernel(const __bf16* __restrict__ X1,
                                                         __bf16* __restrict__ X2) {
  int g = blockIdx.x * 256 + threadIdx.x;   // B*64*M threads
  int b = g / (64 * M_);
  int r = g % (64 * M_);
  int c = r / M_;
  int m = r % M_;
  float mx = -INFINITY;
  __bf16 vals[KNN_];
#pragma unroll
  for (int k = 0; k < KNN_; ++k) {
    vals[k] = X1[pidx(b, c, k * M_ + m, 2)];
    mx = fmaxf(mx, (float)vals[k]);
  }
  __bf16 mb = (__bf16)mx;
#pragma unroll
  for (int k = 0; k < KNN_; ++k) {
    int p = k * M_ + m;
    X2[pidx(b, c,      p, 4)] = vals[k];
    X2[pidx(b, c + 64, p, 4)] = mb;
  }
}

// ---------------------------------------------------------------------------
extern "C" void kernel_launch(void* const* d_in, const int* in_sizes, int n_in,
                              void* d_out, int out_size, void* d_ws, size_t ws_size,
                              hipStream_t stream) {
  (void)in_sizes; (void)n_in; (void)out_size; (void)ws_size;
  const float* xyz   = (const float*)d_in[0];
  const float* mis   = (const float*)d_in[1];
  const float* r1scW = (const float*)d_in[2];
  const float* r1scb = (const float*)d_in[3];
  const float* r1w1  = (const float*)d_in[4];
  const float* r1b1  = (const float*)d_in[5];
  const float* r1w2  = (const float*)d_in[6];
  const float* r1b2  = (const float*)d_in[7];
  const float* r1w3  = (const float*)d_in[8];
  const float* r1b3  = (const float*)d_in[9];
  const float* r2scW = (const float*)d_in[10];
  const float* r2scb = (const float*)d_in[11];
  const float* r2w1  = (const float*)d_in[12];
  const float* r2b1  = (const float*)d_in[13];
  const float* r2w2  = (const float*)d_in[14];
  const float* r2b2  = (const float*)d_in[15];
  const float* r2w3  = (const float*)d_in[16];
  const float* r2b3  = (const float*)d_in[17];
  float* outp = (float*)d_out;

  // Workspace layout (bf16 elements):
  //   feat[B][TPB][1][32][16] | X1[B][TPB][2][32][16] | X2[B][TPB][4][32][16]
  //   then 8 packed weight blocks
  __bf16* feat = (__bf16*)d_ws;
  __bf16* X1   = feat + (size_t)B_ * TPB_ * 1 * 512;
  __bf16* X2   = X1   + (size_t)B_ * TPB_ * 2 * 512;
  __bf16* wp   = X2   + (size_t)B_ * TPB_ * 4 * 512;

  __bf16* p_r1sc = wp;               // 64x32  -> 4*1*512 = 2048
  __bf16* p_r1w1 = p_r1sc + 2048;    // 64x32  -> 2048
  __bf16* p_r1w2 = p_r1w1 + 2048;    // 64x64  -> 4096
  __bf16* p_r1w3 = p_r1w2 + 4096;    // 64x64  -> 4096
  __bf16* p_r2sc = p_r1w3 + 4096;    // 128x128-> 16384
  __bf16* p_r2w1 = p_r2sc + 16384;   // 64x128 -> 8192
  __bf16* p_r2w2 = p_r2w1 + 8192;    // 128x64 -> 8192
  __bf16* p_r2w3 = p_r2w2 + 8192;    // 128x128-> 16384

  // Pack weights into A-fragment order (tiny; L2-resident afterwards)
  pack_weights_kernel<<<(2048  + 255) / 256, 256, 0, stream>>>(r1scW, p_r1sc, 64, 10, 1);
  pack_weights_kernel<<<(2048  + 255) / 256, 256, 0, stream>>>(r1w1,  p_r1w1, 64, 10, 1);
  pack_weights_kernel<<<(4096  + 255) / 256, 256, 0, stream>>>(r1w2,  p_r1w2, 64, 64, 2);
  pack_weights_kernel<<<(4096  + 255) / 256, 256, 0, stream>>>(r1w3,  p_r1w3, 64, 64, 2);
  pack_weights_kernel<<<(16384 + 255) / 256, 256, 0, stream>>>(r2scW, p_r2sc, 128, 128, 4);
  pack_weights_kernel<<<(8192  + 255) / 256, 256, 0, stream>>>(r2w1,  p_r2w1, 64, 128, 4);
  pack_weights_kernel<<<(8192  + 255) / 256, 256, 0, stream>>>(r2w2,  p_r2w2, 128, 64, 2);
  pack_weights_kernel<<<(16384 + 255) / 256, 256, 0, stream>>>(r2w3,  p_r2w3, 128, 128, 4);

  const int nOut = B_ * 128 * M_;
  init_out_kernel<<<(nOut + 255) / 256, 256, 0, stream>>>(outp, nOut);

  knn_feat_kernel<<<B_ * M_, 256, 0, stream>>>(xyz, mis, feat);

  res_block_kernel<1, 64, 64, 64, false><<<B_ * TPB_, 32, 0, stream>>>(
      feat, p_r1sc, r1scb, p_r1w1, r1b1, p_r1w2, r1b2, p_r1w3, r1b3, X1, nullptr);

  concat_max_kernel<<<B_ * 64 * M_ / 256, 256, 0, stream>>>(X1, X2);

  res_block_kernel<4, 64, 128, 128, true><<<B_ * TPB_, 32, 0, stream>>>(
      X2, p_r2sc, r2scb, p_r2w1, r2b1, p_r2w2, r2b2, p_r2w3, r2b3, nullptr, outp);
}
